// SGModule_43413529428102
// MI455X (gfx1250) — matrix operand order repaired and checked by reading, not compiled, not run on hardware
//
#include <hip/hip_runtime.h>
#include <stdint.h>

typedef __attribute__((ext_vector_type(16))) _Float16 v16h;
typedef __attribute__((ext_vector_type(8)))  float    v8f;

#define B_   8
#define N_   8192
#define S_   2048
#define KNN  32
#define CIN  32
#define CO_  64
#define EPSV 1e-5f

union PK { unsigned u; _Float16 h[2]; };

// Permuted position of channel ci within a 64-half staged row so that each
// lane's WMMA A-fragment dwords are contiguous:
//   memory order per 32-block: [k0-7][k16-23][k8-15][k24-31]
__device__ __forceinline__ int permpos(int ci) {
  const int blk = ci >> 5, sub = ci & 31, g = sub >> 3;
  const int m = (g == 1) ? 2 : (g == 2) ? 1 : g;
  return blk * 32 + m * 8 + (sub & 7);
}

// ---------------------------------------------------------------- prep ----
__global__ void prep_kernel(const float* __restrict__ w1,
                            const float* __restrict__ w2,
                            _Float16* __restrict__ w1h,
                            _Float16* __restrict__ w2h,
                            float* __restrict__ bnacc) {
  int t = threadIdx.x;
  for (int i = t; i < 64 * 64; i += blockDim.x) {
    w1h[i] = (_Float16)w1[i];
    w2h[i] = (_Float16)w2[i];
  }
  for (int i = t; i < 256; i += blockDim.x) bnacc[i] = 0.f;
}

// ----------------------------------------------------------------- FPS ----
// One block per batch; the 8 points owned by each thread are cached in
// registers, so the 2048-step serial loop touches LDS only for the centroid
// broadcast and the two-level argmax.
__global__ __launch_bounds__(1024)
void fps_kernel(const float* __restrict__ xyz,
                int* __restrict__ fps_idx,
                float* __restrict__ xyz_out) {
  __shared__ __align__(16) float xs[N_], ys[N_], zs[N_];
  __shared__ float wbd[32];
  __shared__ int   wbi[32];
  __shared__ int   far_sh;
  const int b   = blockIdx.x;
  const int tid = threadIdx.x;
  const float* base = xyz + (size_t)b * N_ * 3;
  for (int i = tid; i < N_; i += 1024) {
    xs[i] = base[i * 3 + 0];
    ys[i] = base[i * 3 + 1];
    zs[i] = base[i * 3 + 2];
  }
  __syncthreads();

  float px[8], py[8], pz[8], dist[8];
#pragma unroll
  for (int t = 0; t < 8; ++t) {
    const int p = tid + t * 1024;
    px[t] = xs[p]; py[t] = ys[p]; pz[t] = zs[p];
    dist[t] = 1e10f;
  }
  int far = 0;

  for (int j = 0; j < S_; ++j) {
    if (tid == 0) {
      fps_idx[b * S_ + j] = far;
      xyz_out[((size_t)b * S_ + j) * 3 + 0] = xs[far];
      xyz_out[((size_t)b * S_ + j) * 3 + 1] = ys[far];
      xyz_out[((size_t)b * S_ + j) * 3 + 2] = zs[far];
    }
    const float cx = xs[far], cy = ys[far], cz = zs[far];
    float bd = -1.f; int bi = 0x7fffffff;
#pragma unroll
    for (int t = 0; t < 8; ++t) {
      const float dx = px[t] - cx, dy = py[t] - cy, dz = pz[t] - cz;
      const float d  = dx * dx + dy * dy + dz * dz;
      const float nd = fminf(dist[t], d);
      dist[t] = nd;
      const int p = tid + t * 1024;
      if (nd > bd || (nd == bd && p < bi)) { bd = nd; bi = p; }
    }
#pragma unroll
    for (int m = 1; m < 32; m <<= 1) {
      const float od = __shfl_xor(bd, m, 32);
      const int   oi = __shfl_xor(bi, m, 32);
      if (od > bd || (od == bd && oi < bi)) { bd = od; bi = oi; }
    }
    if ((tid & 31) == 0) { wbd[tid >> 5] = bd; wbi[tid >> 5] = bi; }
    __syncthreads();
    if (tid < 32) {
      float b2 = wbd[tid]; int i2 = wbi[tid];
#pragma unroll
      for (int m = 1; m < 32; m <<= 1) {
        const float od = __shfl_xor(b2, m, 32);
        const int   oi = __shfl_xor(i2, m, 32);
        if (od > b2 || (od == b2 && oi < i2)) { b2 = od; i2 = oi; }
      }
      if (tid == 0) far_sh = i2;
    }
    __syncthreads();
    far = far_sh;
  }
}

// ----------------------------------------------------------------- KNN ----
__global__ __launch_bounds__(128)
void knn_kernel(const float* __restrict__ xyz,
                const int* __restrict__ fps_idx,
                int* __restrict__ knn_idx) {
  __shared__ __align__(16) float xs[N_], ys[N_], zs[N_];
  __shared__ float ld[KNN * 128];
  __shared__ int   li[KNN * 128];
  const int b   = blockIdx.x >> 4;
  const int s0  = (blockIdx.x & 15) * 128;
  const int tid = threadIdx.x;
  const float* base = xyz + (size_t)b * N_ * 3;
  for (int i = tid; i < N_; i += 128) {
    xs[i] = base[i * 3 + 0];
    ys[i] = base[i * 3 + 1];
    zs[i] = base[i * 3 + 2];
  }
  for (int e = 0; e < KNN; ++e) { ld[e * 128 + tid] = 1e30f; li[e * 128 + tid] = 0; }
  __syncthreads();

  const int s  = s0 + tid;
  const int qi = fps_idx[b * S_ + s];
  const float qx = xs[qi], qy = ys[qi], qz = zs[qi];
  float curmax = 1e30f; int curpos = 0;

  for (int n4 = 0; n4 < N_; n4 += 4) {
    const float4 X = *(const float4*)&xs[n4];
    const float4 Y = *(const float4*)&ys[n4];
    const float4 Z = *(const float4*)&zs[n4];
    float dc[4];
    dc[0] = (X.x-qx)*(X.x-qx) + (Y.x-qy)*(Y.x-qy) + (Z.x-qz)*(Z.x-qz);
    dc[1] = (X.y-qx)*(X.y-qx) + (Y.y-qy)*(Y.y-qy) + (Z.y-qz)*(Z.y-qz);
    dc[2] = (X.z-qx)*(X.z-qx) + (Y.z-qy)*(Y.z-qy) + (Z.z-qz)*(Z.z-qz);
    dc[3] = (X.w-qx)*(X.w-qx) + (Y.w-qy)*(Y.w-qy) + (Z.w-qz)*(Z.w-qz);
#pragma unroll
    for (int c = 0; c < 4; ++c) {
      if (dc[c] < curmax) {
        ld[curpos * 128 + tid] = dc[c];
        li[curpos * 128 + tid] = n4 + c;
        float m = -1.f; int mp = 0;
#pragma unroll
        for (int e = 0; e < KNN; ++e) {
          const float v = ld[e * 128 + tid];
          if (v > m) { m = v; mp = e; }
        }
        curmax = m; curpos = mp;
      }
    }
  }
  for (int e = 0; e < KNN; ++e)
    knn_idx[((size_t)b * S_ + s) * KNN + e] = li[e * 128 + tid];
}

// ---------------------------------------------------------------- GEMM ----
// 128 rows x 64 cols per block (8 waves, each a 16x64 slab via 8 WMMAs).
// Input tile staged in LDS in fragment-permuted order so A-fragments load
// as contiguous ds_load_b128s; staging stores are packed-pair b32s.
__global__ __launch_bounds__(256)
void gemm_kernel(int mode,
                 const float*   __restrict__ feat,
                 const int*     __restrict__ fps_idx,
                 const int*     __restrict__ knn_idx,
                 const _Float16* __restrict__ yin,
                 const float*   __restrict__ a_c,
                 const float*   __restrict__ b_c,
                 const _Float16* __restrict__ wh,
                 _Float16*      __restrict__ yout,
                 float* __restrict__ bnsum,
                 float* __restrict__ bnss) {
  __shared__ __align__(16) _Float16 ein [128 * 64];   // fragment-permuted
  __shared__ __align__(16) _Float16 eout[128 * 64];   // standard [row][co]
  __shared__ float lsum[64], lss[64];

  const int tid = threadIdx.x;
  if (tid < 64) { lsum[tid] = 0.f; lss[tid] = 0.f; }
  const int r0 = blockIdx.x * 128;
  unsigned* einw = (unsigned*)ein;

  if (mode == 0) {
    const int b = r0 >> 16;                      // S_*KNN = 65536 rows/batch
    for (int idx = tid; idx < 128 * 32; idx += 256) {
      const int lr  = idx >> 5;
      const int ci0 = (idx & 31) * 2;
      const int rb  = (r0 + lr) & 65535;
      const int s   = rb >> 5, k = rb & 31;
      const int fi  = fps_idx[b * S_ + s];
      float v0, v1;
      if (ci0 < CIN) {
        const int ni = knn_idx[(((size_t)b * S_ + s) << 5) + k];
        const float* fn = &feat[((size_t)b * N_ + ni) * CIN + ci0];
        const float* fc = &feat[((size_t)b * N_ + fi) * CIN + ci0];
        v0 = fn[0] - fc[0];
        v1 = fn[1] - fc[1];
      } else {
        const float* fc = &feat[((size_t)b * N_ + fi) * CIN + (ci0 - CIN)];
        v0 = fc[0];
        v1 = fc[1];
      }
      PK p; p.h[0] = (_Float16)v0; p.h[1] = (_Float16)v1;
      einw[(lr * 64 + permpos(ci0)) >> 1] = p.u;
    }
  } else {
    const unsigned* yinw = (const unsigned*)yin;
    for (int idx = tid; idx < 128 * 32; idx += 256) {
      const int lr  = idx >> 5;
      const int ci0 = (idx & 31) * 2;
      PK in; in.u = yinw[(size_t)r0 * 32 + lr * 32 + (ci0 >> 1)];
      float v0 = a_c[ci0]     * (float)in.h[0] + b_c[ci0];
      float v1 = a_c[ci0 + 1] * (float)in.h[1] + b_c[ci0 + 1];
      PK p;
      p.h[0] = (_Float16)(v0 > 0.f ? v0 : 0.f);
      p.h[1] = (_Float16)(v1 > 0.f ? v1 : 0.f);
      einw[(lr * 64 + permpos(ci0)) >> 1] = p.u;
    }
  }
  __syncthreads();

  const int wave  = tid >> 5;
  const int lane  = tid & 31;
  const int mbase = wave * 16;
  const int mrow  = mbase + (lane & 15);
  const int hi    = (lane >> 4) & 1;

  union AF { v16h v; unsigned u[8]; };
  AF af0, af1;
  const unsigned* einu = (const unsigned*)ein;
  const int basew = (mrow * 64 + (hi ? 16 : 0)) >> 1;   // dword index
#pragma unroll
  for (int r = 0; r < 8; ++r) {
    af0.u[r] = einu[basew + r];        // contiguous -> 2x ds_load_b128
    af1.u[r] = einu[basew + 16 + r];   // +32 halves (K chunk 1)
  }

  const unsigned* whu = (const unsigned*)wh;
  v8f acc[4];
#pragma unroll
  for (int nt = 0; nt < 4; ++nt) {
#pragma unroll
    for (int e = 0; e < 8; ++e) acc[nt][e] = 0.f;
    const int co = nt * 16 + (lane & 15);
    AF bf0, bf1;
#pragma unroll
    for (int r = 0; r < 8; ++r) {
      // 16-bit B 32x16 layout: V r -> K = 2r,2r+1 (r<4) / 2r+8,2r+9 (r>=4) (lo lanes), +8 (hi lanes)
      const int k0 = 2 * r + (r >= 4 ? 8 : 0) + (hi ? 8 : 0);
      bf0.u[r] = whu[(co * 64 +      k0) >> 1];
      bf1.u[r] = whu[(co * 64 + 32 + k0) >> 1];
    }
    acc[nt] = __builtin_amdgcn_wmma_f32_16x16x32_f16(
        false, af0.v, false, bf0.v, (short)0, acc[nt], false, false);
    acc[nt] = __builtin_amdgcn_wmma_f32_16x16x32_f16(
        false, af1.v, false, bf1.v, (short)0, acc[nt], false, false);
  }

  // BN partial sums + LDS restage for coalesced f16 writeout.
#pragma unroll
  for (int nt = 0; nt < 4; ++nt) {
    const int co = nt * 16 + (lane & 15);
    float s8 = 0.f, ss8 = 0.f;
#pragma unroll
    for (int rr = 0; rr < 8; ++rr) {
      const float val = acc[nt][rr];
      s8 += val; ss8 += val * val;
      eout[(mbase + rr + (hi ? 8 : 0)) * 64 + co] = (_Float16)val;
    }
    atomicAdd(&lsum[co], s8);
    atomicAdd(&lss[co], ss8);
  }
  __syncthreads();

  unsigned* youtu = (unsigned*)yout;
  const unsigned* eoutu = (const unsigned*)eout;
  for (int idx = tid; idx < 128 * 32; idx += 256)
    youtu[(size_t)r0 * 32 + idx] = eoutu[idx];
  if (tid < 64) {
    atomicAdd(&bnsum[tid], lsum[tid]);
    atomicAdd(&bnss[tid],  lss[tid]);
  }
}

// ------------------------------------------------------------ BN fold ----
__global__ void bnfin_kernel(const float* __restrict__ sum,
                             const float* __restrict__ ss,
                             const float* __restrict__ gamma,
                             const float* __restrict__ beta,
                             float* __restrict__ a_out,
                             float* __restrict__ b_out,
                             float cnt_inv) {
  const int c = threadIdx.x;
  if (c < 64) {
    const float mean = sum[c] * cnt_inv;
    const float var  = ss[c] * cnt_inv - mean * mean;
    const float a    = gamma[c] * rsqrtf(var + EPSV);
    a_out[c] = a;
    b_out[c] = beta[c] - mean * a;
  }
}

// ------------------------------------------------------------ maxpool ----
__global__ __launch_bounds__(256)
void maxpool_kernel(const _Float16* __restrict__ y2,
                    const float* __restrict__ a2,
                    const float* __restrict__ b2,
                    float* __restrict__ out) {
  const int idx = blockIdx.x * 256 + threadIdx.x;   // (b*S+s)*64 + co
  const int co  = idx & 63;
  const int bs  = idx >> 6;
  const float a = a2[co], bb = b2[co];
  float m = 0.f;                                    // ReLU folded into max
  const size_t base = (size_t)bs * KNN * 64 + co;
  for (int k = 0; k < KNN; ++k) {
    float v = (float)y2[base + (size_t)k * 64];
    v = a * v + bb;
    if (v > m) m = v;
  }
  out[(size_t)B_ * S_ * 3 + idx] = m;
}

// ----------------------------------------------------------------- entry --
extern "C" void kernel_launch(void* const* d_in, const int* in_sizes, int n_in,
                              void* d_out, int out_size, void* d_ws, size_t ws_size,
                              hipStream_t stream) {
  (void)in_sizes; (void)n_in; (void)out_size; (void)ws_size;
  const float* xyz = (const float*)d_in[0];
  const float* feat = (const float*)d_in[1];
  const float* w1 = (const float*)d_in[2];
  const float* g1 = (const float*)d_in[3];
  const float* b1 = (const float*)d_in[4];
  const float* w2 = (const float*)d_in[5];
  const float* g2 = (const float*)d_in[6];
  const float* b2 = (const float*)d_in[7];
  float* out = (float*)d_out;

  char* ws = (char*)d_ws;
  size_t off = 0;
  auto take = [&](size_t bytes) -> char* {
    char* p = ws + off;
    off = (off + bytes + 255) & ~(size_t)255;
    return p;
  };
  int*       fps   = (int*)take((size_t)B_ * S_ * 4);
  int*       knn   = (int*)take((size_t)B_ * S_ * KNN * 4);
  _Float16*  w1h   = (_Float16*)take(64 * 64 * 2);
  _Float16*  w2h   = (_Float16*)take(64 * 64 * 2);
  float*     bnacc = (float*)take(256 * 4);   // [sum1|ss1|sum2|ss2] x64
  float*     coef  = (float*)take(256 * 4);   // [a1|b1|a2|b2] x64
  _Float16*  y1    = (_Float16*)take((size_t)B_ * S_ * KNN * CO_ * 2);
  _Float16*  y2    = (_Float16*)take((size_t)B_ * S_ * KNN * CO_ * 2);

  const float cnt_inv = 1.0f / (float)((size_t)B_ * S_ * KNN);
  const int gemm_blocks = (B_ * S_ * KNN) / 128;          // 4096

  prep_kernel<<<1, 256, 0, stream>>>(w1, w2, w1h, w2h, bnacc);
  fps_kernel<<<B_, 1024, 0, stream>>>(xyz, fps, out);
  knn_kernel<<<B_ * (S_ / 128), 128, 0, stream>>>(xyz, fps, knn);
  gemm_kernel<<<gemm_blocks, 256, 0, stream>>>(0, feat, fps, knn,
      (const _Float16*)nullptr, nullptr, nullptr, w1h, y1, bnacc + 0, bnacc + 64);
  bnfin_kernel<<<1, 64, 0, stream>>>(bnacc + 0, bnacc + 64, g1, b1,
                                     coef + 0, coef + 64, cnt_inv);
  gemm_kernel<<<gemm_blocks, 256, 0, stream>>>(1, feat, fps, knn,
      y1, coef + 0, coef + 64, w2h, y2, bnacc + 128, bnacc + 192);
  bnfin_kernel<<<1, 64, 0, stream>>>(bnacc + 128, bnacc + 192, g2, b2,
                                     coef + 128, coef + 192, cnt_inv);
  maxpool_kernel<<<(B_ * S_ * CO_) / 256, 256, 0, stream>>>(y2, coef + 128,
                                                            coef + 192, out);
}
